// Longformer_67869073211747
// MI455X (gfx1250) — compile-verified, matrix operand order checked by reading, not run on hardware
//
#include <hip/hip_runtime.h>
#include <math.h>

#define S_LEN   4096
#define DMODEL  768
#define NHEAD   12
#define DHEAD   64
#define FDIM    3072
#define NLAYER  12
#define NGLOB   8
#define GSTRIDE 512
#define ODIM    128

typedef __attribute__((ext_vector_type(16))) __bf16 v16bf;
typedef __attribute__((ext_vector_type(8)))  float  v8f;

union BF16x16 { v16bf v; unsigned short u[16]; unsigned int w[8]; };

__device__ __forceinline__ unsigned short f2bf(float f) {
  unsigned int x = __float_as_uint(f);
  unsigned int r = x + 0x7fffu + ((x >> 16) & 1u);
  return (unsigned short)(r >> 16);
}
__device__ __forceinline__ unsigned int pack2bf(float a, float b) {
  return (unsigned int)f2bf(a) | ((unsigned int)f2bf(b) << 16);
}
__device__ __forceinline__ float bf2f(unsigned short h) {
  return __uint_as_float(((unsigned int)h) << 16);
}
__device__ __forceinline__ float bflo(unsigned int w) {
  return __uint_as_float(w << 16);
}
__device__ __forceinline__ float bfhi(unsigned int w) {
  return __uint_as_float(w & 0xffff0000u);
}

// CDNA5 async global->LDS copy (no VGPR round trip, tracked by ASYNCcnt).
// Generic pointers to LDS carry the LDS byte offset in addr[31:0] (ISA 10.2).
__device__ __forceinline__ void async_ld_b128(void* lds, const void* g) {
  unsigned lds32 = (unsigned)(uintptr_t)lds;
  asm volatile("global_load_async_to_lds_b128 %0, %1, off"
               :: "v"(lds32), "v"(g) : "memory");
}
__device__ __forceinline__ void wait_async0() {
#if __has_builtin(__builtin_amdgcn_s_wait_asynccnt)
  __builtin_amdgcn_s_wait_asynccnt(0);
#else
  asm volatile("s_wait_asynccnt 0" ::: "memory");
#endif
}

// Load one 16x32 bf16 A/B fragment (16-bit WMMA VGPR layout):
// lane<16 : elems 0-7 -> K 0..7,  elems 8-15 -> K 16..23
// lane>=16: elems 0-7 -> K 8..15, elems 8-15 -> K 24..31
// p must be 4B aligned; kb_ is 0 or 8 (even -> aligned u32 pair loads).
__device__ __forceinline__ void load_frag16x32(const unsigned short* p, int kb_,
                                               BF16x16& f) {
#pragma unroll
  for (int j = 0; j < 4; j++) {
    f.w[j]     = *(const unsigned int*)(p + kb_ + 2 * j);
    f.w[4 + j] = *(const unsigned int*)(p + kb_ + 16 + 2 * j);
  }
}

// ---------------------------------------------------------------- embed + LN
__global__ __launch_bounds__(256, 1) void embed_ln_kernel(
    const int* __restrict__ x, const float* __restrict__ wemb,
    const float* __restrict__ pemb, const float* __restrict__ g,
    const float* __restrict__ b, float* __restrict__ h,
    unsigned short* __restrict__ hb) {
  __shared__ float red[256];
  const int r = blockIdx.x, tid = threadIdx.x;
  const int tok = x[r];
  float v0 = wemb[(size_t)tok * DMODEL + tid]       + pemb[(size_t)r * DMODEL + tid];
  float v1 = wemb[(size_t)tok * DMODEL + tid + 256] + pemb[(size_t)r * DMODEL + tid + 256];
  float v2 = wemb[(size_t)tok * DMODEL + tid + 512] + pemb[(size_t)r * DMODEL + tid + 512];
  red[tid] = v0 + v1 + v2;
  __syncthreads();
  for (int st = 128; st > 0; st >>= 1) { if (tid < st) red[tid] += red[tid + st]; __syncthreads(); }
  const float mean = red[0] * (1.0f / DMODEL);
  __syncthreads();
  const float d0 = v0 - mean, d1 = v1 - mean, d2 = v2 - mean;
  red[tid] = d0 * d0 + d1 * d1 + d2 * d2;
  __syncthreads();
  for (int st = 128; st > 0; st >>= 1) { if (tid < st) red[tid] += red[tid + st]; __syncthreads(); }
  const float rstd = rsqrtf(red[0] * (1.0f / DMODEL) + 1e-5f);
  float y0 = d0 * rstd * g[tid]       + b[tid];
  float y1 = d1 * rstd * g[tid + 256] + b[tid + 256];
  float y2 = d2 * rstd * g[tid + 512] + b[tid + 512];
  size_t base = (size_t)r * DMODEL;
  h[base + tid] = y0;        hb[base + tid] = f2bf(y0);
  h[base + tid + 256] = y1;  hb[base + tid + 256] = f2bf(y1);
  h[base + tid + 512] = y2;  hb[base + tid + 512] = f2bf(y2);
}

// ------------------------------------------------------- residual + LN
__global__ __launch_bounds__(256, 1) void ln_res_kernel(
    const float* __restrict__ hin, const float* __restrict__ delta,
    const float* __restrict__ g, const float* __restrict__ b,
    float* __restrict__ hout, unsigned short* __restrict__ hb) {
  __shared__ float red[256];
  const int r = blockIdx.x, tid = threadIdx.x;
  const size_t base = (size_t)r * DMODEL;
  float v0 = hin[base + tid]       + delta[base + tid];
  float v1 = hin[base + tid + 256] + delta[base + tid + 256];
  float v2 = hin[base + tid + 512] + delta[base + tid + 512];
  red[tid] = v0 + v1 + v2;
  __syncthreads();
  for (int st = 128; st > 0; st >>= 1) { if (tid < st) red[tid] += red[tid + st]; __syncthreads(); }
  const float mean = red[0] * (1.0f / DMODEL);
  __syncthreads();
  const float d0 = v0 - mean, d1 = v1 - mean, d2 = v2 - mean;
  red[tid] = d0 * d0 + d1 * d1 + d2 * d2;
  __syncthreads();
  for (int st = 128; st > 0; st >>= 1) { if (tid < st) red[tid] += red[tid + st]; __syncthreads(); }
  const float rstd = rsqrtf(red[0] * (1.0f / DMODEL) + 1e-5f);
  float y0 = d0 * rstd * g[tid]       + b[tid];
  float y1 = d1 * rstd * g[tid + 256] + b[tid + 256];
  float y2 = d2 * rstd * g[tid + 512] + b[tid + 512];
  hout[base + tid] = y0;        hb[base + tid] = f2bf(y0);
  hout[base + tid + 256] = y1;  hb[base + tid + 256] = f2bf(y1);
  hout[base + tid + 512] = y2;  hb[base + tid + 512] = f2bf(y2);
}

// ----------------------------------------------------------------- GEMM
// Y[M=4096 x N] = Xb(bf16)[M x K] @ W(f32->bf16)[K x N] + bias ; opt GELU
// Block tile 128x128, 8 waves (4 M x 2 N), wave tile 32x64, K-step 32.
// Double-buffered LDS. A tile: async global->LDS (no VGPR round trip).
// B tile: register-staged with early f32->bf16 packing (needs conversion).
// One barrier per 8 WMMA; next tile's copies overlap current WMMA phase.
__global__ __launch_bounds__(256, 1) void gemm_bias_kernel(
    const unsigned short* __restrict__ Xb, const float* __restrict__ W,
    const float* __restrict__ bias, float* __restrict__ Yf,
    unsigned short* __restrict__ Yb, int N, int K, int act) {
  __shared__ unsigned short As[2][128][40];  // [buf][m][k], 80B rows (16B aligned)
  __shared__ unsigned short Bs[2][128][40];  // [buf][n][k]
  const int tid = threadIdx.x;
  const int lane = tid & 31, wave = tid >> 5;
  const int wm = wave & 3, wn = wave >> 2;
  const int lanehi = (lane >> 4) & 1, lo = lane & 15;
  const int bm = blockIdx.x * 128, bn = blockIdx.y * 128;
  const int kb_ = lanehi ? 8 : 0;

  v8f acc[2][4] = {};
  unsigned int tbp[8];                         // 4 slots x 2 packed bf16 pairs

  auto stage_a_async = [&](int k0, int buf) {  // A: 128x32 bf16 = 512 b128 copies
#pragma unroll
    for (int i = 0; i < 2; i++) {
      int e = tid + 256 * i;
      int m = e >> 2, kc = (e & 3) * 8;
      async_ld_b128(&As[buf][m][kc], &Xb[(size_t)(bm + m) * K + k0 + kc]);
    }
  };
  auto load_b = [&](int k0) {                  // B: 32x128 f32 = 1024 float4
#pragma unroll
    for (int i = 0; i < 4; i++) {
      int e = tid + 256 * i;
      int kk = e >> 5, n4 = (e & 31) * 4;
      float4 w = *(const float4*)&W[(size_t)(k0 + kk) * N + bn + n4];
      tbp[2 * i]     = pack2bf(w.x, w.y);      // convert early: 8 live regs not 16
      tbp[2 * i + 1] = pack2bf(w.z, w.w);
    }
  };
  auto store_b = [&](int buf) {
#pragma unroll
    for (int i = 0; i < 4; i++) {
      int e = tid + 256 * i;
      int kk = e >> 5, n4 = (e & 31) * 4;
      unsigned int p0 = tbp[2 * i], p1 = tbp[2 * i + 1];
      Bs[buf][n4 + 0][kk] = (unsigned short)p0;
      Bs[buf][n4 + 1][kk] = (unsigned short)(p0 >> 16);
      Bs[buf][n4 + 2][kk] = (unsigned short)p1;
      Bs[buf][n4 + 3][kk] = (unsigned short)(p1 >> 16);
    }
  };

  stage_a_async(0, 0);
  load_b(0);
  store_b(0);
  wait_async0();
  int cur = 0;
  const int nk = K >> 5;
  for (int kt = 0; kt < nk; kt++) {
    __syncthreads();                           // staged tile visible
    const bool more = (kt + 1) < nk;
    if (more) {
      stage_a_async((kt + 1) << 5, cur ^ 1);   // async copy overlaps WMMA below
      load_b((kt + 1) << 5);
    }

    BF16x16 a0, a1;
    load_frag16x32(&As[cur][wm * 32 + lo][0], kb_, a0);
    load_frag16x32(&As[cur][wm * 32 + 16 + lo][0], kb_, a1);
#pragma unroll
    for (int nf = 0; nf < 4; nf++) {
      BF16x16 bf;
      load_frag16x32(&Bs[cur][wn * 64 + nf * 16 + lo][0], kb_, bf);
      acc[0][nf] = __builtin_amdgcn_wmma_f32_16x16x32_bf16(
          false, a0.v, false, bf.v, (short)0, acc[0][nf], false, false);
      acc[1][nf] = __builtin_amdgcn_wmma_f32_16x16x32_bf16(
          false, a1.v, false, bf.v, (short)0, acc[1][nf], false, false);
    }
    if (more) store_b(cur ^ 1);                // fill other buffer, no barrier
    wait_async0();                             // our async A writes landed
    cur ^= 1;
  }

#pragma unroll
  for (int mf = 0; mf < 2; mf++)
#pragma unroll
    for (int nf = 0; nf < 4; nf++) {
      const int col = bn + wn * 64 + nf * 16 + lo;
      const float bs = bias[col];
#pragma unroll
      for (int r = 0; r < 8; r++) {
        const int row = bm + wm * 32 + mf * 16 + r + 8 * lanehi;
        float v = acc[mf][nf][r] + bs;
        if (act) v = 0.5f * v * (1.0f + erff(v * 0.70710678118f));
        if (Yf) Yf[(size_t)row * N + col] = v;
        if (Yb) Yb[(size_t)row * N + col] = f2bf(v);
      }
    }
}

// --------------------------------------------- band (sliding-window) attn
// block = (64 queries, 1 head); 4 waves * 16 queries; flash online softmax.
// 18 band chunks of 32 keys covering [q0-256, q0+63+256] + 1 global-key chunk.
// K/V chunk staging via async global->LDS, double-buffered; out-of-range keys
// use clamped addresses (their probabilities are masked to zero anyway).
__global__ __launch_bounds__(128, 1) void band_attn_kernel(
    const unsigned short* __restrict__ qbuf, const unsigned short* __restrict__ kbuf,
    const unsigned short* __restrict__ vbuf, const int* __restrict__ xmask,
    unsigned short* __restrict__ ctxb) {
  __shared__ unsigned short Ks[2][32][72];
  __shared__ unsigned short Vs[2][32][72];
  __shared__ float Ssc[4][16][32];
  __shared__ unsigned short Ps[4][16][40];
  __shared__ float mst[4][16], lst[4][16], rfac[4][16];
  __shared__ int kok[2][32];

  const int tid = threadIdx.x;
  const int lane = tid & 31, wave = tid >> 5;
  const int lanehi = (lane >> 4) & 1, lo = lane & 15;
  const int head = blockIdx.y;
  const int q0 = blockIdx.x * 64;
  const int qw = q0 + wave * 16;
  const int col0 = head * DHEAD;
  const int kb_ = lanehi ? 8 : 0;
  const int kstart = q0 - 256;
  const float scale = 0.125f;

  // Q fragments (16 queries x 64 dims -> two 16x32 frags), straight from global bf16
  BF16x16 qf[2];
  {
    const unsigned short* qp = qbuf + (size_t)(qw + lo) * DMODEL + col0;
    load_frag16x32(qp, kb_, qf[0]);
    load_frag16x32(qp + 32, kb_, qf[1]);
  }

  auto stage_kv_async = [&](int cn, int buf) {
#pragma unroll
    for (int i = 0; i < 2; i++) {
      int slot = tid + 128 * i;                // 0..255
      int key = slot >> 3, d8 = (slot & 7) * 8;
      int kabs;
      if (cn < 18) kabs = kstart + cn * 32 + key;
      else         kabs = key * GSTRIDE;
      int kc = kabs < 0 ? 0 : (kabs >= S_LEN ? S_LEN - 1 : kabs);  // clamp; masked later
      size_t base = (size_t)kc * DMODEL + col0 + d8;
      async_ld_b128(&Ks[buf][key][d8], &kbuf[base]);
      async_ld_b128(&Vs[buf][key][d8], &vbuf[base]);
    }
    if (tid < 32) {                            // per-key validity (query-independent)
      bool ok;
      if (cn < 18) {
        int kabs = kstart + cn * 32 + tid;
        bool inr = (kabs >= 0) && (kabs < S_LEN);
        ok = inr && ((kabs & (GSTRIDE - 1)) != 0) && (xmask[inr ? kabs : 0] > 0);
      } else {
        ok = tid < NGLOB;                      // global key slots are force-valid
      }
      kok[buf][tid] = ok ? 1 : 0;
    }
  };

  if (lane < 16) { mst[wave][lane] = -1e30f; lst[wave][lane] = 0.0f; }
  v8f cacc[4] = {};

  stage_kv_async(0, 0);
  wait_async0();
  int cur = 0;
  for (int c = 0; c <= 18; c++) {
    __syncthreads();                           // staged chunk visible
    const bool more = c < 18;
    if (more) stage_kv_async(c + 1, cur ^ 1);  // async copy overlaps both WMMA phases

    // ---- scores: 16 queries x 32 keys via WMMA
    v8f s0 = {}, s1 = {};
    {
      BF16x16 bk;
      load_frag16x32(&Ks[cur][lo][0], kb_, bk);        // keys 0..15, d 0..31
      s0 = __builtin_amdgcn_wmma_f32_16x16x32_bf16(false, qf[0].v, false, bk.v, (short)0, s0, false, false);
      load_frag16x32(&Ks[cur][lo][0] + 32, kb_, bk);   // keys 0..15, d 32..63
      s0 = __builtin_amdgcn_wmma_f32_16x16x32_bf16(false, qf[1].v, false, bk.v, (short)0, s0, false, false);
      load_frag16x32(&Ks[cur][16 + lo][0], kb_, bk);   // keys 16..31
      s1 = __builtin_amdgcn_wmma_f32_16x16x32_bf16(false, qf[0].v, false, bk.v, (short)0, s1, false, false);
      load_frag16x32(&Ks[cur][16 + lo][0] + 32, kb_, bk);
      s1 = __builtin_amdgcn_wmma_f32_16x16x32_bf16(false, qf[1].v, false, bk.v, (short)0, s1, false, false);
    }
#pragma unroll
    for (int r = 0; r < 8; r++) {
      int row = r + 8 * lanehi;
      Ssc[wave][row][lo] = s0[r];
      Ssc[wave][row][16 + lo] = s1[r];
    }
    __syncthreads();

    // ---- online softmax update (lane r handles row r)
    if (lane < 16) {
      const int row = lane;
      const int q = qw + row;
      const float m_old = mst[wave][row];
      const float l_old = lst[wave][row];
      float mx = m_old;
#pragma unroll
      for (int j = 0; j < 32; j++) {
        float s = Ssc[wave][row][j] * scale;
        bool valid = (kok[cur][j] != 0);
        if (c < 18) {
          int diff = kstart + c * 32 + j - q;
          valid = valid && (diff <= 256) && (diff >= -256);
        }
        if (valid) mx = fmaxf(mx, s);
      }
      const float rf = __expf(m_old - mx);
      float sum = 0.0f;
#pragma unroll
      for (int j = 0; j < 32; j++) {
        float s = Ssc[wave][row][j] * scale;
        bool valid = (kok[cur][j] != 0);
        if (c < 18) {
          int diff = kstart + c * 32 + j - q;
          valid = valid && (diff <= 256) && (diff >= -256);
        }
        float p = valid ? __expf(s - mx) : 0.0f;
        Ps[wave][row][j] = f2bf(p);
        sum += p;
      }
      mst[wave][row] = mx;
      lst[wave][row] = l_old * rf + sum;
      rfac[wave][row] = rf;
    }
    __syncthreads();

    // ---- rescale context accumulators, then ctx += P @ V
#pragma unroll
    for (int r = 0; r < 8; r++) {
      float f = rfac[wave][r + 8 * lanehi];
#pragma unroll
      for (int nt = 0; nt < 4; nt++) cacc[nt][r] *= f;
    }
    BF16x16 pf;
    load_frag16x32(&Ps[wave][lo][0], kb_, pf);         // P: 16 queries x 32 keys
#pragma unroll
    for (int nt = 0; nt < 4; nt++) {
      BF16x16 vf;
      const int d = nt * 16 + lo;
#pragma unroll
      for (int i = 0; i < 16; i++) {
        int key = (i < 8) ? (kb_ + i) : (kb_ + 16 + (i - 8));
        vf.u[i] = Vs[cur][key][d];
      }
      cacc[nt] = __builtin_amdgcn_wmma_f32_16x16x32_bf16(
          false, pf.v, false, vf.v, (short)0, cacc[nt], false, false);
    }
    wait_async0();                             // our async K/V writes landed
    cur ^= 1;
  }

  // ---- normalize and write context (bf16, feeds Wo GEMM)
#pragma unroll
  for (int r = 0; r < 8; r++) {
    const int row = r + 8 * lanehi;
    const float l = lst[wave][row];
    const float inv = (l > 0.0f) ? (1.0f / l) : 0.0f;
    const size_t gbase = (size_t)(qw + row) * DMODEL + col0;
#pragma unroll
    for (int nt = 0; nt < 4; nt++)
      ctxb[gbase + nt * 16 + lo] = f2bf(cacc[nt][r] * inv);
  }
}

// --------------------------------------------- global-token full attention
// grid = NHEAD blocks; 8 global queries, full attention over S keys.
__global__ __launch_bounds__(256, 1) void global_attn_kernel(
    const float* __restrict__ h, const float* __restrict__ Wqg,
    const float* __restrict__ bqg, const unsigned short* __restrict__ kgb,
    const unsigned short* __restrict__ vgb, const int* __restrict__ xmask,
    unsigned short* __restrict__ ctxb, float* __restrict__ gsc) {
  __shared__ float qgs[NGLOB][DHEAD];
  __shared__ float red[256];
  __shared__ float mx[NGLOB], sm[NGLOB];
  __shared__ float part[4][NGLOB][DHEAD];
  const int head = blockIdx.x, tid = threadIdx.x;
  const int col0 = head * DHEAD;
  const float scale = 0.125f;
  float* gs = gsc + (size_t)head * NGLOB * S_LEN;

  // qg projection for this head's 8 global rows
  for (int o = tid; o < NGLOB * DHEAD; o += 256) {
    int i = o >> 6, d = o & 63;
    const float* hr = h + (size_t)(i * GSTRIDE) * DMODEL;
    float acc = bqg[col0 + d];
    for (int cc = 0; cc < DMODEL; cc++) acc += hr[cc] * Wqg[(size_t)cc * DMODEL + col0 + d];
    qgs[i][d] = acc;
  }
  __syncthreads();

  // phase 1: scores + local max
  float lmax[NGLOB];
#pragma unroll
  for (int i = 0; i < NGLOB; i++) lmax[i] = -1e30f;
  for (int s = tid; s < S_LEN; s += 256) {
    float kr[DHEAD];
#pragma unroll
    for (int d8 = 0; d8 < DHEAD; d8 += 8) {
      uint4 raw = *(const uint4*)&kgb[(size_t)s * DMODEL + col0 + d8];
      kr[d8 + 0] = bflo(raw.x); kr[d8 + 1] = bfhi(raw.x);
      kr[d8 + 2] = bflo(raw.y); kr[d8 + 3] = bfhi(raw.y);
      kr[d8 + 4] = bflo(raw.z); kr[d8 + 5] = bfhi(raw.z);
      kr[d8 + 6] = bflo(raw.w); kr[d8 + 7] = bfhi(raw.w);
    }
    const bool valid = (xmask[s] > 0) || ((s & (GSTRIDE - 1)) == 0);
#pragma unroll
    for (int i = 0; i < NGLOB; i++) {
      float acc = 0.0f;
#pragma unroll
      for (int d = 0; d < DHEAD; d++) acc += qgs[i][d] * kr[d];
      float sc = valid ? acc * scale : -1e30f;
      gs[(size_t)i * S_LEN + s] = sc;
      lmax[i] = fmaxf(lmax[i], sc);
    }
  }
  for (int i = 0; i < NGLOB; i++) {
    red[tid] = lmax[i];
    __syncthreads();
    for (int st = 128; st > 0; st >>= 1) { if (tid < st) red[tid] = fmaxf(red[tid], red[tid + st]); __syncthreads(); }
    if (tid == 0) mx[i] = red[0];
    __syncthreads();
  }

  // phase 2: exp + sum
  float lsum[NGLOB] = {};
  for (int s = tid; s < S_LEN; s += 256) {
#pragma unroll
    for (int i = 0; i < NGLOB; i++) {
      float p = __expf(gs[(size_t)i * S_LEN + s] - mx[i]);
      gs[(size_t)i * S_LEN + s] = p;
      lsum[i] += p;
    }
  }
  for (int i = 0; i < NGLOB; i++) {
    red[tid] = lsum[i];
    __syncthreads();
    for (int st = 128; st > 0; st >>= 1) { if (tid < st) red[tid] += red[tid + st]; __syncthreads(); }
    if (tid == 0) sm[i] = red[0];
    __syncthreads();
  }

  // phase 3: ctx = P @ Vg
  const int d = tid & 63, qtr = tid >> 6;
  float acc[NGLOB] = {};
  for (int s = qtr * (S_LEN / 4); s < (qtr + 1) * (S_LEN / 4); s++) {
    float vv = bf2f(vgb[(size_t)s * DMODEL + col0 + d]);
#pragma unroll
    for (int i = 0; i < NGLOB; i++) acc[i] += gs[(size_t)i * S_LEN + s] * vv;
  }
#pragma unroll
  for (int i = 0; i < NGLOB; i++) part[qtr][i][d] = acc[i];
  __syncthreads();
  if (tid < NGLOB * DHEAD) {
    int i = tid >> 6, dd = tid & 63;
    float c = part[0][i][dd] + part[1][i][dd] + part[2][i][dd] + part[3][i][dd];
    float denom = sm[i];
    float o = (denom > 0.0f) ? c / denom : 0.0f;
    ctxb[(size_t)(i * GSTRIDE) * DMODEL + col0 + dd] = f2bf(o);  // overwrite band ctx
  }
}

// ------------------------------------------------------------ output head
__global__ __launch_bounds__(128, 1) void out_proj_kernel(
    const float* __restrict__ h, const float* __restrict__ ow,
    const float* __restrict__ ob, float* __restrict__ out) {
  const int i = blockIdx.x, o = threadIdx.x;
  const float* hr = h + (size_t)(i * GSTRIDE) * DMODEL;  // CLS row of window i
  float acc = ob[o];
  for (int c = 0; c < DMODEL; c++) acc += hr[c] * ow[(size_t)c * ODIM + o];
  out[i * ODIM + o] = acc;
}

// ================================================================== launch
extern "C" void kernel_launch(void* const* d_in, const int* in_sizes, int n_in,
                              void* d_out, int out_size, void* d_ws, size_t ws_size,
                              hipStream_t stream) {
  (void)in_sizes; (void)n_in; (void)out_size; (void)ws_size;
  const int*   x     = (const int*)d_in[0];
  const int*   xmask = (const int*)d_in[1];
  const float* wemb  = (const float*)d_in[2];
  const float* pemb  = (const float*)d_in[3];
  const float* embg  = (const float*)d_in[4];
  const float* embb  = (const float*)d_in[5];
  const float* Wq  = (const float*)d_in[6];  const float* bq  = (const float*)d_in[7];
  const float* Wk  = (const float*)d_in[8];  const float* bk  = (const float*)d_in[9];
  const float* Wv  = (const float*)d_in[10]; const float* bv  = (const float*)d_in[11];
  const float* Wo  = (const float*)d_in[12]; const float* bo  = (const float*)d_in[13];
  const float* Wqg = (const float*)d_in[14]; const float* bqg = (const float*)d_in[15];
  const float* Wkg = (const float*)d_in[16]; const float* bkg = (const float*)d_in[17];
  const float* Wvg = (const float*)d_in[18]; const float* bvg = (const float*)d_in[19];
  const float* l1g = (const float*)d_in[20]; const float* l1b = (const float*)d_in[21];
  const float* W1  = (const float*)d_in[22]; const float* b1  = (const float*)d_in[23];
  const float* W2  = (const float*)d_in[24]; const float* b2  = (const float*)d_in[25];
  const float* l2g = (const float*)d_in[26]; const float* l2b = (const float*)d_in[27];
  const float* outw = (const float*)d_in[28]; const float* outb = (const float*)d_in[29];

  char* ws = (char*)d_ws;
  size_t off = 0;
  auto alloc = [&](size_t bytes) -> void* {
    void* p = ws + off;
    off += (bytes + 255) & ~(size_t)255;
    return p;
  };
  float*          h    = (float*)         alloc((size_t)S_LEN * DMODEL * 4);
  unsigned short* hb   = (unsigned short*)alloc((size_t)S_LEN * DMODEL * 2);
  unsigned short* qb   = (unsigned short*)alloc((size_t)S_LEN * DMODEL * 2);
  unsigned short* kbb  = (unsigned short*)alloc((size_t)S_LEN * DMODEL * 2);
  unsigned short* vbb  = (unsigned short*)alloc((size_t)S_LEN * DMODEL * 2);
  unsigned short* kgb  = (unsigned short*)alloc((size_t)S_LEN * DMODEL * 2);
  unsigned short* vgb  = (unsigned short*)alloc((size_t)S_LEN * DMODEL * 2);
  unsigned short* ctxb = (unsigned short*)alloc((size_t)S_LEN * DMODEL * 2);
  unsigned short* ffb  = (unsigned short*)alloc((size_t)S_LEN * FDIM * 2);
  float*          attn = (float*)         alloc((size_t)S_LEN * DMODEL * 4);
  float*          gsc  = (float*)         alloc((size_t)NHEAD * NGLOB * S_LEN * 4);

  embed_ln_kernel<<<S_LEN, 256, 0, stream>>>(x, wemb, pemb, embg, embb, h, hb);

  const dim3 g768(S_LEN / 128, DMODEL / 128);
  const dim3 g3072(S_LEN / 128, FDIM / 128);
  const dim3 gband(S_LEN / 64, NHEAD);

  for (int l = 0; l < NLAYER; l++) {
    const size_t wDD = (size_t)l * DMODEL * DMODEL;
    const size_t wDF = (size_t)l * DMODEL * FDIM;
    const size_t vD  = (size_t)l * DMODEL;
    const size_t vF  = (size_t)l * FDIM;

    gemm_bias_kernel<<<g768, 256, 0, stream>>>(hb, Wq + wDD, bq + vD, nullptr, qb,  DMODEL, DMODEL, 0);
    gemm_bias_kernel<<<g768, 256, 0, stream>>>(hb, Wk + wDD, bk + vD, nullptr, kbb, DMODEL, DMODEL, 0);
    gemm_bias_kernel<<<g768, 256, 0, stream>>>(hb, Wv + wDD, bv + vD, nullptr, vbb, DMODEL, DMODEL, 0);
    gemm_bias_kernel<<<g768, 256, 0, stream>>>(hb, Wkg + wDD, bkg + vD, nullptr, kgb, DMODEL, DMODEL, 0);
    gemm_bias_kernel<<<g768, 256, 0, stream>>>(hb, Wvg + wDD, bvg + vD, nullptr, vgb, DMODEL, DMODEL, 0);

    band_attn_kernel<<<gband, 128, 0, stream>>>(qb, kbb, vbb, xmask, ctxb);
    global_attn_kernel<<<NHEAD, 256, 0, stream>>>(h, Wqg + wDD, bqg + vD, kgb, vgb, xmask, ctxb, gsc);

    gemm_bias_kernel<<<g768, 256, 0, stream>>>(ctxb, Wo + wDD, bo + vD, attn, nullptr, DMODEL, DMODEL, 0);
    ln_res_kernel<<<S_LEN, 256, 0, stream>>>(h, attn, l1g + vD, l1b + vD, h, hb);

    gemm_bias_kernel<<<g3072, 256, 0, stream>>>(hb, W1 + wDF, b1 + vF, nullptr, ffb, FDIM, DMODEL, 1);
    gemm_bias_kernel<<<g768, 256, 0, stream>>>(ffb, W2 + wDF, b2 + vD, attn, nullptr, DMODEL, FDIM, 0);
    ln_res_kernel<<<S_LEN, 256, 0, stream>>>(h, attn, l2g + vD, l2b + vD, h, hb);
  }

  out_proj_kernel<<<NGLOB, ODIM, 0, stream>>>(h, outw, outb, (float*)d_out);
}
